// MambaMixer_44109314129924
// MI455X (gfx1250) — compile-verified, matrix operand order checked by reading, not run on hardware
//
#include <hip/hip_runtime.h>
#include <hip/hip_bf16.h>

// ---------------- Problem constants ----------------
#define BATCH  2
#define SEQ    2048
#define HID    2048
#define INTER  4096
#define NSTATE 16
#define DTRANK 128
#define KCONV  4
#define GNORM  64

#define BL       (BATCH * SEQ)          // 4096 token rows
#define PROJW    (2 * INTER)            // 8192
#define NSBC     (DTRANK + 2 * NSTATE)  // 160 fused ts/B/C width
#define NSBC_PAD 256                    // padded to grid tile multiple (zero rows)

typedef __bf16 bf16;
typedef __attribute__((ext_vector_type(16))) __bf16 v16bf;
typedef __attribute__((ext_vector_type(8)))  float  v8f;

__device__ __forceinline__ float silu_f(float x) {
    return x / (1.0f + __expf(-x));
}

// Low 32 bits of a generic pointer to LDS == byte offset from LDS base
__device__ __forceinline__ unsigned lds_addr32(const void* p) {
    return (unsigned)(unsigned long long)p;
}

// CDNA5 async global->LDS copy, 16B per lane, tracked by ASYNCcnt
__device__ __forceinline__ void async_copy_b128(unsigned ldsOff, const void* gptr) {
    asm volatile("global_load_async_to_lds_b128 %0, %1, off"
                 :: "v"(ldsOff), "v"(gptr) : "memory");
}

// ---------------- f32 -> bf16 cast ----------------
__global__ __launch_bounds__(256) void cast_bf16_kernel(const float* __restrict__ src,
                                                        bf16* __restrict__ dst, int n) {
    int i = blockIdx.x * 256 + threadIdx.x;
    if (i < n) dst[i] = (bf16)src[i];
}

// Fused [W_step(128); W_B(16); W_C(16)] weight, zero-padded to 256 rows so the
// GEMM's async B-tile loads never need a guard.
__global__ __launch_bounds__(256) void build_wsbc_kernel(const float* __restrict__ Wstep,
                                                         const float* __restrict__ WB,
                                                         const float* __restrict__ WC,
                                                         bf16* __restrict__ W) {
    int i = blockIdx.x * 256 + threadIdx.x;   // over 256*4096
    if (i >= NSBC_PAD * INTER) return;
    int r = i / INTER, c = i - r * INTER;
    float v = 0.0f;
    if (r < DTRANK)                 v = Wstep[r * INTER + c];
    else if (r < DTRANK + NSTATE)   v = WB[(r - DTRANK) * INTER + c];
    else if (r < NSBC)              v = WC[(r - DTRANK - NSTATE) * INTER + c];
    W[i] = (bf16)v;
}

// ts columns (0..127) of tsbc -> bf16 for the dt GEMM
__global__ __launch_bounds__(256) void ts_cast_kernel(const float* __restrict__ tsbc,
                                                      bf16* __restrict__ tsb) {
    int i = blockIdx.x * 256 + threadIdx.x;   // over 4096*128
    if (i >= BL * DTRANK) return;
    int r = i >> 7, c = i & 127;
    tsb[i] = (bf16)tsbc[r * NSBC + c];
}

// ---------------- WMMA bf16 GEMM: C[M,N] = A[M,K] * B[N,K]^T ----------------
// Double-buffered async(global->LDS) pipeline unrolled by 2 K-tiles so the
// buffer index is a compile-time constant in each half-step.
// Requirements: M % 128 == 0, K % 64 == 0 (even tile count), B has
// >= gridDim.x*128 readable rows (zero-padded); C stores guarded by true N.
#define TILE_K     32
#define LDS_STRIDE 40   // 32 bf16 + pad; 80B row stride keeps 16B alignment

__device__ __forceinline__ v16bf lds_frag(const bf16* __restrict__ p, int lane) {
    // lanes 0-15: K chunks {0..7, 16..23}; lanes 16-31: {8..15, 24..31}
    const int hi = (lane >> 4) << 3;
    union { v16bf v; uint4 q[2]; } f;
    f.q[0] = *reinterpret_cast<const uint4*>(p + hi);
    f.q[1] = *reinterpret_cast<const uint4*>(p + 16 + hi);
    return f.v;
}

__global__ __launch_bounds__(256) void wmma_gemm_bf16_kernel(
        const bf16* __restrict__ A, const bf16* __restrict__ B,
        float* __restrict__ C, int M, int N, int K) {
    __shared__ __align__(16) bf16 As[2][128 * LDS_STRIDE];
    __shared__ __align__(16) bf16 Bs[2][128 * LDS_STRIDE];

    const int tid  = threadIdx.x;
    const int lane = tid & 31;          // wave32
    const int wave = tid >> 5;          // 8 waves
    const int wm   = wave & 3;          // 4 waves x 32 rows
    const int wn   = wave >> 2;         // 2 waves x 64 cols
    const int mBase = blockIdx.y * 128;
    const int nBase = blockIdx.x * 128;

    // Staging addresses: 512 16B-chunks per tile, 2 per thread (rows r, r+64)
    const int rowS = tid >> 2;
    const int koS  = (tid & 3) << 3;
    const bf16* gA0 = A + (size_t)(mBase + rowS) * K + koS;
    const bf16* gA1 = A + (size_t)(mBase + rowS + 64) * K + koS;
    const bf16* gB0 = B + (size_t)(nBase + rowS) * K + koS;
    const bf16* gB1 = B + (size_t)(nBase + rowS + 64) * K + koS;
    unsigned lA0[2], lA1[2], lB0[2], lB1[2];
#pragma unroll
    for (int p = 0; p < 2; ++p) {
        lA0[p] = lds_addr32(&As[p][rowS * LDS_STRIDE + koS]);
        lA1[p] = lds_addr32(&As[p][(rowS + 64) * LDS_STRIDE + koS]);
        lB0[p] = lds_addr32(&Bs[p][rowS * LDS_STRIDE + koS]);
        lB1[p] = lds_addr32(&Bs[p][(rowS + 64) * LDS_STRIDE + koS]);
    }

    v8f acc[2][4];
    const v8f vzero = {0.f, 0.f, 0.f, 0.f, 0.f, 0.f, 0.f, 0.f};
#pragma unroll
    for (int i = 0; i < 2; ++i)
#pragma unroll
        for (int j = 0; j < 4; ++j) acc[i][j] = vzero;

    auto issue = [&](int p, int tile) {
        const int kn = tile * TILE_K;
        async_copy_b128(lA0[p], gA0 + kn);
        async_copy_b128(lA1[p], gA1 + kn);
        async_copy_b128(lB0[p], gB0 + kn);
        async_copy_b128(lB1[p], gB1 + kn);
    };

    const int lrow = lane & 15;
    auto compute = [&](int p) {
        v16bf afrag[2], bfrag[4];
#pragma unroll
        for (int i = 0; i < 2; ++i)
            afrag[i] = lds_frag(&As[p][(wm * 32 + i * 16 + lrow) * LDS_STRIDE], lane);
#pragma unroll
        for (int j = 0; j < 4; ++j)
            bfrag[j] = lds_frag(&Bs[p][(wn * 64 + j * 16 + lrow) * LDS_STRIDE], lane);
#pragma unroll
        for (int i = 0; i < 2; ++i)
#pragma unroll
            for (int j = 0; j < 4; ++j)
                acc[i][j] = __builtin_amdgcn_wmma_f32_16x16x32_bf16(
                    false, afrag[i], false, bfrag[j],
                    (short)0, acc[i][j], false, false);
    };

    // Prologue: two tiles in flight (8 outstanding async ops per wave)
    const int ntiles = K / TILE_K;   // even by contract
    issue(0, 0);
    issue(1, 1);

    for (int t = 0; t < ntiles; t += 2) {
        const bool more = (t + 2) < ntiles;

        // --- tile t in buffer 0 (oldest 4 outstanding) ---
        asm volatile("s_wait_asynccnt 0x4" ::: "memory");
        __syncthreads();
        compute(0);
        __syncthreads();               // all waves done reading buffer 0
        if (more) issue(0, t + 2);

        // --- tile t+1 in buffer 1 ---
        if (more) asm volatile("s_wait_asynccnt 0x4" ::: "memory");
        else      asm volatile("s_wait_asynccnt 0x0" ::: "memory");
        __syncthreads();
        compute(1);
        __syncthreads();               // all waves done reading buffer 1
        if (more) issue(1, t + 3);
    }

    // Store: VGPR r -> M = r (+8 for lanes 16-31), N = lane&15
    const int nCol = lane & 15;
    const int mAdd = (lane >> 4) << 3;
#pragma unroll
    for (int i = 0; i < 2; ++i) {
        const int rBase = mBase + wm * 32 + i * 16;
#pragma unroll
        for (int j = 0; j < 4; ++j) {
            const int col = nBase + wn * 64 + j * 16 + nCol;
            if (col < N) {
#pragma unroll
                for (int r = 0; r < 8; ++r)
                    C[(size_t)(rBase + r + mAdd) * N + col] = acc[i][j][r];
            }
        }
    }
}

// ---------------- depthwise conv(K=4) + bias + SiLU ----------------
__global__ __launch_bounds__(256) void conv_silu_kernel(
        const float* __restrict__ proj, const float* __restrict__ conv_w,
        const float* __restrict__ conv_b, float* __restrict__ hs_act,
        bf16* __restrict__ hsT) {
    const int bl = blockIdx.y;
    const int b  = bl / SEQ;
    const int l  = bl - b * SEQ;
    const int e  = blockIdx.x * 256 + threadIdx.x;

    float acc = conv_b[e];
#pragma unroll
    for (int j = 0; j < KCONV; ++j) {
        int ls = l - (KCONV - 1) + j;
        if (ls >= 0)
            acc += conv_w[e * KCONV + j] * proj[(size_t)(b * SEQ + ls) * PROJW + e];
    }
    float s = silu_f(acc);
    hs_act[(size_t)bl * INTER + e] = s;
    hsT[(size_t)bl * INTER + e]    = (bf16)s;
}

// ---------------- selective scan ----------------
__global__ __launch_bounds__(256) void scan_kernel(
        const float* __restrict__ dts,    // [BL, INTER]
        const float* __restrict__ dt_b,   // [INTER]
        const float* __restrict__ A_log,  // [INTER, 16]
        const float* __restrict__ hs_act, // [BL, INTER]
        const float* __restrict__ tsbc,   // [BL, 160]: B at 128.., C at 144..
        float* __restrict__ y) {          // [BL, INTER]
    const int idx = blockIdx.x * 256 + threadIdx.x;   // 0..8191
    const int b = idx >> 12;
    const int e = idx & (INTER - 1);

    float A[NSTATE], h[NSTATE];
#pragma unroll
    for (int n = 0; n < NSTATE; ++n) {
        A[n] = -__expf(A_log[e * NSTATE + n]);
        h[n] = 0.0f;
    }
    const float db = dt_b[e];

    for (int l = 0; l < SEQ; ++l) {
        const size_t row = (size_t)(b * SEQ + l);
        float x  = dts[row * INTER + e] + db;
        float dt = (x > 20.0f) ? x : log1pf(__expf(x));   // softplus
        float u  = hs_act[row * INTER + e];
        float du = dt * u;
        const float* Bt = tsbc + row * NSBC + DTRANK;
        const float* Ct = Bt + NSTATE;
        float yv = 0.0f;
#pragma unroll
        for (int n = 0; n < NSTATE; ++n) {
            h[n] = h[n] * __expf(dt * A[n]) + du * Bt[n];
            yv += h[n] * Ct[n];
        }
        y[row * INTER + e] = yv;
    }
}

// ---------------- RMS group-norm(64) + D-skip + SiLU gate -> bf16 ----------------
// One wave32 per 64-channel group (2 channels per lane): pure shuffle reduce,
// no LDS, no barriers. Block = 8 waves = 512 channels of one token.
__global__ __launch_bounds__(256) void gn_gate_kernel(
        const float* __restrict__ y, const float* __restrict__ hs_act,
        const float* __restrict__ proj, const float* __restrict__ D_skip,
        bf16* __restrict__ act) {
    const int bl   = blockIdx.x >> 3;                 // token index
    const int wave = threadIdx.x >> 5;
    const int lane = threadIdx.x & 31;
    const int grp  = (blockIdx.x & 7) * 8 + wave;     // group 0..63
    const int e    = grp * GNORM + lane * 2;          // even channel pair

    const float2 v2 = *reinterpret_cast<const float2*>(y + (size_t)bl * INTER + e);
    float ss = v2.x * v2.x + v2.y * v2.y;
#pragma unroll
    for (int s = 16; s >= 1; s >>= 1) ss += __shfl_xor(ss, s, 32);
    const float rinv = rsqrtf(ss * (1.0f / (float)GNORM) + 1e-6f);

    const float2 hs2 = *reinterpret_cast<const float2*>(hs_act + (size_t)bl * INTER + e);
    const float2 g2  = *reinterpret_cast<const float2*>(proj + (size_t)bl * PROJW + INTER + e);
    const float2 d2  = *reinterpret_cast<const float2*>(D_skip + e);

    union { unsigned u; bf16 h[2]; } pk;
    pk.h[0] = (bf16)((v2.x * rinv + hs2.x * d2.x) * silu_f(g2.x));
    pk.h[1] = (bf16)((v2.y * rinv + hs2.y * d2.y) * silu_f(g2.y));
    *reinterpret_cast<unsigned*>(act + (size_t)bl * INTER + e) = pk.u;
}

// ---------------- host orchestration ----------------
extern "C" void kernel_launch(void* const* d_in, const int* in_sizes, int n_in,
                              void* d_out, int out_size, void* d_ws, size_t ws_size,
                              hipStream_t stream) {
    (void)in_sizes; (void)n_in; (void)out_size; (void)ws_size;

    const float* hidden = (const float*)d_in[0];
    const float* W_in   = (const float*)d_in[1];
    const float* conv_w = (const float*)d_in[2];
    const float* conv_b = (const float*)d_in[3];
    const float* W_step = (const float*)d_in[4];
    const float* W_B    = (const float*)d_in[5];
    const float* W_C    = (const float*)d_in[6];
    const float* dt_w   = (const float*)d_in[7];
    const float* dt_b   = (const float*)d_in[8];
    const float* A_log  = (const float*)d_in[9];
    const float* D_skip = (const float*)d_in[10];
    const float* W_out  = (const float*)d_in[11];
    float* out = (float*)d_out;

    char* ws = (char*)d_ws;
    size_t off = 0;
    auto alloc = [&](size_t bytes) -> void* {
        void* p = ws + off;
        off += (bytes + 255) & ~(size_t)255;
        return p;
    };

    bf16*  hs_bf   = (bf16*) alloc((size_t)BL * HID * 2);
    bf16*  win_bf  = (bf16*) alloc((size_t)PROJW * HID * 2);
    bf16*  wout_bf = (bf16*) alloc((size_t)HID * INTER * 2);
    bf16*  wsbc_bf = (bf16*) alloc((size_t)NSBC_PAD * INTER * 2);  // zero-padded
    bf16*  dtw_bf  = (bf16*) alloc((size_t)INTER * DTRANK * 2);
    bf16*  ts_bf   = (bf16*) alloc((size_t)BL * DTRANK * 2);
    float* proj    = (float*)alloc((size_t)BL * PROJW * 4);
    float* hs_act  = (float*)alloc((size_t)BL * INTER * 4);
    bf16*  hsT_bf  = (bf16*) alloc((size_t)BL * INTER * 2);
    float* tsbc    = (float*)alloc((size_t)BL * NSBC * 4);
    float* dts     = (float*)alloc((size_t)BL * INTER * 4);
    float* ybuf    = (float*)alloc((size_t)BL * INTER * 4);
    bf16*  act_bf  = hsT_bf;   // dead after ts/B/C GEMM -> reuse

    // 1) casts / weight prep
    cast_bf16_kernel<<<(BL * HID) / 256, 256, 0, stream>>>(hidden, hs_bf, BL * HID);
    cast_bf16_kernel<<<(PROJW * HID) / 256, 256, 0, stream>>>(W_in, win_bf, PROJW * HID);
    cast_bf16_kernel<<<(HID * INTER) / 256, 256, 0, stream>>>(W_out, wout_bf, HID * INTER);
    cast_bf16_kernel<<<(INTER * DTRANK) / 256, 256, 0, stream>>>(dt_w, dtw_bf, INTER * DTRANK);
    build_wsbc_kernel<<<(NSBC_PAD * INTER) / 256, 256, 0, stream>>>(W_step, W_B, W_C, wsbc_bf);

    // 2) in-projection: [4096,8192] = hs[4096,2048] x W_in[8192,2048]^T
    wmma_gemm_bf16_kernel<<<dim3(PROJW / 128, BL / 128), 256, 0, stream>>>(
        hs_bf, win_bf, proj, BL, PROJW, HID);

    // 3) depthwise conv + SiLU
    conv_silu_kernel<<<dim3(INTER / 256, BL), 256, 0, stream>>>(
        proj, conv_w, conv_b, hs_act, hsT_bf);

    // 4) fused ts/B/C: [4096,160] = hsT[4096,4096] x Wsbc[160(pad 256),4096]^T
    wmma_gemm_bf16_kernel<<<dim3(NSBC_PAD / 128, BL / 128), 256, 0, stream>>>(
        hsT_bf, wsbc_bf, tsbc, BL, NSBC, INTER);

    // 5) ts -> bf16
    ts_cast_kernel<<<(BL * DTRANK) / 256, 256, 0, stream>>>(tsbc, ts_bf);

    // 6) dt expansion: [4096,4096] = ts[4096,128] x dt_w[4096,128]^T  (4 K-tiles)
    wmma_gemm_bf16_kernel<<<dim3(INTER / 128, BL / 128), 256, 0, stream>>>(
        ts_bf, dtw_bf, dts, BL, INTER, DTRANK);

    // 7) selective scan
    scan_kernel<<<(BATCH * INTER) / 256, 256, 0, stream>>>(
        dts, dt_b, A_log, hs_act, tsbc, ybuf);

    // 8) group norm + skip + gate -> bf16
    gn_gate_kernel<<<BL * 8, 256, 0, stream>>>(
        ybuf, hs_act, proj, D_skip, act_bf);

    // 9) out-projection: [4096,2048] = act[4096,4096] x W_out[2048,4096]^T
    wmma_gemm_bf16_kernel<<<dim3(HID / 128, BL / 128), 256, 0, stream>>>(
        act_bf, wout_bf, out, BL, HID, INTER);
}
